// Simple_channel_mixing_module_17815524344444
// MI455X (gfx1250) — compile-verified, matrix operand order checked by reading, not run on hardware
//
#include <hip/hip_runtime.h>

#define H    128
#define CCH  16
#define NB   16384
#define ROWS 32
#define LDH  136   // padded row stride in halves for f16 tiles (128 + 8)
#define TLD  132   // padded row stride in floats for T tile

typedef __attribute__((ext_vector_type(16))) _Float16 v16h;
typedef __attribute__((ext_vector_type(8)))  _Float16 v8h;
typedef __attribute__((ext_vector_type(4)))  _Float16 v4h;
typedef __attribute__((ext_vector_type(8)))  float    v8f;

// Load a 16x32 f16 WMMA fragment (A-layout == B-layout for X @ W.T):
// lane l holds row (l&15), 16 contiguous halves starting at K-offset (l>>4)*16.
__device__ __forceinline__ v16h load_frag(const _Float16* base, int lane) {
  const int r  = lane & 15;
  const int hs = lane >> 4;
  const _Float16* p = base + r * LDH + hs * 16;
  v8h lo = *reinterpret_cast<const v8h*>(p);
  v8h hi = *reinterpret_cast<const v8h*>(p + 8);
  return __builtin_shufflevector(lo, hi, 0,1,2,3,4,5,6,7,8,9,10,11,12,13,14,15);
}

__global__ __launch_bounds__(256, 1)
void channel_mix_kernel(const float* __restrict__ x,
                        const float* __restrict__ R,
                        const float* __restrict__ Qm,
                        float* __restrict__ out) {
  // f16 LDS tiles (CDNA5: 320 KB/WGP; this uses ~229 KB)
  __shared__ __align__(16) _Float16 ldsA[H * LDH];              // R - Q
  __shared__ __align__(16) _Float16 ldsQ[H * LDH];              // Q
  __shared__ __align__(16) _Float16 ldsX[CCH * ROWS * LDH];     // x tile, per channel
  __shared__ __align__(16) _Float16 ldsS[ROWS * LDH];           // channel sum
  __shared__ __align__(16) float    ldsT[ROWS * TLD];           // T = S @ Q.T (f32)

  const int tid  = threadIdx.x;
  const int lane = tid & 31;
  const int wave = tid >> 5;
  const int row0 = blockIdx.x * ROWS;

  // ---- 1. Weights: convert R-Q and Q to f16 in LDS (hits L2 after first WG)
  for (int i = tid; i < H * H; i += 256) {
    const int r = i >> 7;
    const int c = i & 127;
    const float q = Qm[i];
    ldsQ[r * LDH + c] = (_Float16)q;
    ldsA[r * LDH + c] = (_Float16)(R[i] - q);
  }

  // ---- 2. x tile: 32 rows x 2048 f32, float4-coalesced, convert to f16
  {
    const float4* xv = reinterpret_cast<const float4*>(x + (size_t)row0 * (CCH * H));
    for (int i = tid; i < ROWS * 512; i += 256) {
      const float4 v = xv[i];
      const int r  = i >> 9;            // row in tile
      const int cc = (i & 511) << 2;    // col 0..2044 (stays inside one channel)
      const int ch = cc >> 7;
      const int k  = cc & 127;
      v4h h = { (_Float16)v.x, (_Float16)v.y, (_Float16)v.z, (_Float16)v.w };
      *reinterpret_cast<v4h*>(&ldsX[((ch * ROWS) + r) * LDH + k]) = h;
    }
  }
  __syncthreads();

  // ---- 3. S = sum over channels (f32 accumulate, stored f16)
  for (int i = tid; i < ROWS * H; i += 256) {
    const int r = i >> 7;
    const int k = i & 127;
    float s = 0.f;
    #pragma unroll
    for (int ch = 0; ch < CCH; ++ch)
      s += (float)ldsX[((ch * ROWS) + r) * LDH + k];
    ldsS[r * LDH + k] = (_Float16)s;
  }
  __syncthreads();

  // ---- 4. T = S @ Q.T  (2 rowblocks x 8 col-subtiles = 16 WMMA tiles, 2/wave)
  for (int t = wave; t < 16; t += 8) {
    const int rb = t & 1;
    const int ns = t >> 1;
    v8f acc = {};
    #pragma unroll
    for (int ks = 0; ks < 4; ++ks) {
      v16h a = load_frag(&ldsS[(rb * 16) * LDH + ks * 32], lane);
      v16h b = load_frag(&ldsQ[(ns * 16) * LDH + ks * 32], lane);
      acc = __builtin_amdgcn_wmma_f32_16x16x32_f16(false, a, false, b,
                                                   (short)0, acc, false, false);
    }
    const int n  = lane & 15;
    const int hs = lane >> 4;
    #pragma unroll
    for (int q = 0; q < 8; ++q)
      ldsT[(rb * 16 + q + 8 * hs) * TLD + ns * 16 + n] = acc[q];
  }
  __syncthreads();

  // ---- 5. out_c = x_c @ (R-Q).T + T   (each wave: 4 (rowblock, channel) tiles)
  for (int j = 0; j < 4; ++j) {
    const int idx = (wave << 2) + j;
    const int rb  = idx & 1;
    const int ch  = idx >> 1;
    const _Float16* xbase = &ldsX[((ch * ROWS) + rb * 16) * LDH];

    v16h afr[4];
    #pragma unroll
    for (int ks = 0; ks < 4; ++ks)
      afr[ks] = load_frag(xbase + ks * 32, lane);

    const int n  = lane & 15;
    const int hs = lane >> 4;
    #pragma unroll
    for (int ns = 0; ns < 8; ++ns) {
      v8f acc;
      #pragma unroll
      for (int q = 0; q < 8; ++q)
        acc[q] = ldsT[(rb * 16 + q + 8 * hs) * TLD + ns * 16 + n];
      #pragma unroll
      for (int ks = 0; ks < 4; ++ks) {
        v16h b = load_frag(&ldsA[(ns * 16) * LDH + ks * 32], lane);
        acc = __builtin_amdgcn_wmma_f32_16x16x32_f16(false, afr[ks], false, b,
                                                     (short)0, acc, false, false);
      }
      float* ob = out + (size_t)(row0 + rb * 16 + 8 * hs) * (CCH * H)
                      + ch * H + ns * 16 + n;
      #pragma unroll
      for (int q = 0; q < 8; ++q)
        ob[(size_t)q * (CCH * H)] = acc[q];
    }
  }
}

extern "C" void kernel_launch(void* const* d_in, const int* in_sizes, int n_in,
                              void* d_out, int out_size, void* d_ws, size_t ws_size,
                              hipStream_t stream) {
  const float* x = (const float*)d_in[0];
  const float* R = (const float*)d_in[1];
  const float* Q = (const float*)d_in[2];
  float* out = (float*)d_out;
  (void)in_sizes; (void)n_in; (void)out_size; (void)d_ws; (void)ws_size;

  dim3 grid(NB / ROWS);   // 512 workgroups, each owns a 32-row tile
  dim3 block(256);        // 8 wave32
  channel_mix_kernel<<<grid, block, 0, stream>>>(x, R, Q, out);
}
